// WindowAttention_19344532701890
// MI455X (gfx1250) — compile-verified, matrix operand order checked by reading, not run on hardware
//
#include <hip/hip_runtime.h>
#include <hip/hip_bf16.h>

#define NWIN  4096
#define NTOK  64
#define CDIM  512
#define NHEAD 16
#define HDIM  32
#define NTHREADS 512          // 16 wave32s per window/workgroup

// LDS strides (elements). All fragment loads are 16B-aligned b128s.
#define XS  520   // xs / outs row stride (ushort)  -> 1040B = 65*16
#define QS  32    // qs / ks row stride (ushort)    -> 64B
#define VTS 72    // vt row stride (ushort)         -> 144B = 9*16
#define SS  68    // S row stride (float)           -> 272B = 17*16
#define PS  72    // P row stride (ushort)          -> 144B = 9*16

typedef __attribute__((ext_vector_type(16))) __bf16 bf16x16;
typedef __attribute__((ext_vector_type(8)))  float  f32x8;

union ABfrag { unsigned int u[8]; bf16x16 v; };

__device__ __forceinline__ unsigned short f2bf(float f) {
  unsigned int u = __float_as_uint(f);
  u += 0x7fffu + ((u >> 16) & 1u);           // round-to-nearest-even
  return (unsigned short)(u >> 16);
}
__device__ __forceinline__ unsigned int packbf2(float lo, float hi) {
  return (unsigned int)f2bf(lo) | ((unsigned int)f2bf(hi) << 16);
}

__device__ __forceinline__ f32x8 wmma_bf16(bf16x16 a, bf16x16 b, f32x8 c) {
  // v_wmma_f32_16x16x32_bf16 : D = A(16x32) * B(32x16) + C
  return __builtin_amdgcn_wmma_f32_16x16x32_bf16(
      false, a, false, b, (short)0, c, false, false);
}

// 16-bit A/B fragment from LDS (ISA 7.12.2): lane m=lane&15, K runs
// [kb, kb+7] and [kb+16, kb+23] with kb = (lane>=16)*8. Two ds_load_b128.
__device__ __forceinline__ bf16x16 lds_frag(const unsigned short* base,
                                            int stride, int lane) {
  const int m  = lane & 15;
  const int kb = (lane >> 4) << 3;
  const unsigned short* p = base + m * stride + kb;
  ABfrag r;
  *(uint4*)(&r.u[0]) = *(const uint4*)(p);
  *(uint4*)(&r.u[4]) = *(const uint4*)(p + 16);
  return r.v;
}

// B fragment from pre-converted bf16 weights in global (L2-resident):
// two 16B runs -> 2x global_load_b128, zero conversion ALU in the hot loop.
__device__ __forceinline__ bf16x16 gmem_wfrag(const unsigned short* base,
                                              int ld, int lane) {
  const int n  = lane & 15;
  const int kb = (lane >> 4) << 3;
  const unsigned short* p = base + (size_t)n * ld + kb;
  ABfrag r;
  *(uint4*)(&r.u[0]) = *(const uint4*)(p);
  *(uint4*)(&r.u[4]) = *(const uint4*)(p + 16);
  return r.v;
}

// ---- one-shot fp32 -> bf16 weight conversion into workspace ----------------
__global__ __launch_bounds__(256)
void convert_w_bf16(const float* __restrict__ src,
                    unsigned short* __restrict__ dst, int n4) {
  int i = blockIdx.x * 256 + threadIdx.x;
  if (i < n4) {
    float4 f = ((const float4*)src)[i];
    uint2 pk;
    pk.x = packbf2(f.x, f.y);
    pk.y = packbf2(f.z, f.w);
    ((uint2*)dst)[i] = pk;
  }
}

__global__ __launch_bounds__(NTHREADS, 1)
void swin_window_attention(const float* __restrict__ x,
                           const float* __restrict__ mask,
                           const unsigned short* __restrict__ qkv_w,   // bf16
                           const float* __restrict__ qkv_b,
                           const unsigned short* __restrict__ proj_w,  // bf16
                           const float* __restrict__ proj_b,
                           const float* __restrict__ bias_table,
                           const int*   __restrict__ rel_index,
                           float* __restrict__ out) {
  __shared__ __align__(16) unsigned short xs  [NTOK * XS];   // x in bf16
  __shared__ __align__(16) unsigned short outs[NTOK * XS];   // per-head O, bf16
  __shared__ __align__(16) unsigned short qs  [NTOK * QS];   // q (scale folded)
  __shared__ __align__(16) unsigned short ks  [NTOK * QS];   // k
  __shared__ __align__(16) unsigned short vt  [HDIM * VTS];  // v transposed
  __shared__ __align__(16) float          Sb  [NTOK * SS];   // scores fp32
  __shared__ __align__(16) unsigned short ps  [NTOK * PS];   // probs bf16

  const int w    = blockIdx.x;
  const int tid  = threadIdx.x;
  const int wave = tid >> 5;
  const int lane = tid & 31;
  const float scale = 0.17677669529663687f;   // 1/sqrt(32)

  // D-fragment (C/D layout) decode used by every epilogue
  const int col = lane & 15;
  const int ro  = (lane >> 4) << 3;

  // ---------------- Stage x tile (64x512 fp32 -> bf16 LDS), read HBM once ---
  {
    const float4* xg = (const float4*)(x + (size_t)w * NTOK * CDIM);
#pragma unroll 4
    for (int i = 0; i < 16; ++i) {
      int f   = i * NTHREADS + tid;   // flat float4 index, row stride = 128
      int row = f >> 7;
      int c4  = f & 127;
      float4 xv = xg[f];
      uint2 pk;
      pk.x = packbf2(xv.x, xv.y);
      pk.y = packbf2(xv.z, xv.w);
      *(uint2*)(&xs[row * XS + (c4 << 2)]) = pk;
    }
  }

  // Hoist this thread's softmax mask segment + bias gather indices
  // (8 lanes per row: r = tid/8, 8 columns each)
  const int r  = tid >> 3;             // softmax row 0..63
  const int c0 = (tid & 7) << 3;       // 8-col segment
  float mreg[8];
  int   bix[8];
  {
    const float* mrow = mask + ((size_t)w * NTOK + r) * NTOK + c0;
    const int*   rrow = rel_index + r * NTOK + c0;
    float4 m0 = *(const float4*)(mrow);
    float4 m1 = *(const float4*)(mrow + 4);
    int4   i0 = *(const int4*)(rrow);
    int4   i1 = *(const int4*)(rrow + 4);
    mreg[0]=m0.x; mreg[1]=m0.y; mreg[2]=m0.z; mreg[3]=m0.w;
    mreg[4]=m1.x; mreg[5]=m1.y; mreg[6]=m1.z; mreg[7]=m1.w;
    bix[0]=i0.x*NHEAD; bix[1]=i0.y*NHEAD; bix[2]=i0.z*NHEAD; bix[3]=i0.w*NHEAD;
    bix[4]=i1.x*NHEAD; bix[5]=i1.y*NHEAD; bix[6]=i1.z*NHEAD; bix[7]=i1.w*NHEAD;
  }
  __syncthreads();

  // =========================== per-head attention ==========================
  for (int h = 0; h < NHEAD; ++h) {
    // ---- QKV projection pass 1: waves 0-7 -> q tiles, waves 8-15 -> k ----
    {
      const int mat01 = wave >> 3;       // 0=q, 1=k (uniform per wave)
      const int tt    = wave & 7;
      const int rt    = tt >> 1;
      const int ct    = tt & 1;
      const int o0    = mat01 * CDIM + h * HDIM + ct * 16;
      f32x8 acc = {0.f,0.f,0.f,0.f,0.f,0.f,0.f,0.f};
      const unsigned short* abase = xs + rt * 16 * XS;
      const unsigned short* wbase = qkv_w + (size_t)o0 * CDIM;
      // pull next head's weight panel toward the WGP (global_prefetch_b8)
      __builtin_prefetch(wbase + (size_t)HDIM * CDIM, 0, 1);
#pragma unroll 4
      for (int kc = 0; kc < CDIM / 32; ++kc) {
        bf16x16 a = lds_frag(abase + kc * 32, XS, lane);
        bf16x16 b = gmem_wfrag(wbase + kc * 32, CDIM, lane);
        acc = wmma_bf16(a, b, acc);
      }
      unsigned short* dst = mat01 ? ks : qs;        // same store formula
      const float fmul = mat01 ? 1.0f : scale;      // fold softmax scale into q
      const float bb = qkv_b[o0 + col];
#pragma unroll
      for (int rr = 0; rr < 8; ++rr) {
        const int tok = rt * 16 + ro + rr;
        dst[tok * QS + ct * 16 + col] = f2bf((acc[rr] + bb) * fmul);
      }
    }
    // ---- QKV projection pass 2: waves 0-7 -> v tiles (transposed store) ----
    if (wave < 8) {
      const int rt = wave >> 1;
      const int ct = wave & 1;
      const int o0 = 2 * CDIM + h * HDIM + ct * 16;
      f32x8 acc = {0.f,0.f,0.f,0.f,0.f,0.f,0.f,0.f};
      const unsigned short* abase = xs + rt * 16 * XS;
      const unsigned short* wbase = qkv_w + (size_t)o0 * CDIM;
#pragma unroll 4
      for (int kc = 0; kc < CDIM / 32; ++kc) {
        bf16x16 a = lds_frag(abase + kc * 32, XS, lane);
        bf16x16 b = gmem_wfrag(wbase + kc * 32, CDIM, lane);
        acc = wmma_bf16(a, b, acc);
      }
      const float bb = qkv_b[o0 + col];
#pragma unroll
      for (int rr = 0; rr < 8; ++rr) {
        const int tok = rt * 16 + ro + rr;
        vt[(ct * 16 + col) * VTS + tok] = f2bf(acc[rr] + bb);
      }
    }
    __syncthreads();

    // ---- S = (q*scale) k^T : 16 tiles, one per wave, single K=32 WMMA ----
    {
      const int nt = wave >> 2;
      const int mt = wave & 3;
      f32x8 acc = {0.f,0.f,0.f,0.f,0.f,0.f,0.f,0.f};
      bf16x16 a = lds_frag(qs + nt * 16 * QS, QS, lane);
      bf16x16 b = lds_frag(ks + mt * 16 * QS, QS, lane);
      acc = wmma_bf16(a, b, acc);
#pragma unroll
      for (int rr = 0; rr < 8; ++rr)
        Sb[(nt * 16 + ro + rr) * SS + mt * 16 + col] = acc[rr];
    }
    __syncthreads();

    // ---- softmax: 8 threads/row, vector S reads, shfl-xor 1/2/4 reduce ----
    {
      float4 s0 = *(const float4*)(&Sb[r * SS + c0]);
      float4 s1 = *(const float4*)(&Sb[r * SS + c0 + 4]);
      float e[8] = {s0.x, s0.y, s0.z, s0.w, s1.x, s1.y, s1.z, s1.w};
      float mx = -3.0e38f;
#pragma unroll
      for (int j = 0; j < 8; ++j) {
        e[j] += bias_table[bix[j] + h] + mreg[j];
        mx = fmaxf(mx, e[j]);
      }
      mx = fmaxf(mx, __shfl_xor(mx, 1, 32));
      mx = fmaxf(mx, __shfl_xor(mx, 2, 32));
      mx = fmaxf(mx, __shfl_xor(mx, 4, 32));
      float sum = 0.f;
#pragma unroll
      for (int j = 0; j < 8; ++j) { e[j] = __expf(e[j] - mx); sum += e[j]; }
      sum += __shfl_xor(sum, 1, 32);
      sum += __shfl_xor(sum, 2, 32);
      sum += __shfl_xor(sum, 4, 32);
      const float inv = 1.0f / sum;
      uint4 pk4;
      pk4.x = packbf2(e[0] * inv, e[1] * inv);
      pk4.y = packbf2(e[2] * inv, e[3] * inv);
      pk4.z = packbf2(e[4] * inv, e[5] * inv);
      pk4.w = packbf2(e[6] * inv, e[7] * inv);
      *(uint4*)(ps + r * PS + c0) = pk4;            // one ds_store_b128
    }
    __syncthreads();

    // ---- O = P V : 8 tiles (4 row x 2 col), waves 0..7, K=64 -> 2 WMMA ----
    if (wave < 8) {
      const int nt = wave >> 1;
      const int dt = wave & 1;
      f32x8 acc = {0.f,0.f,0.f,0.f,0.f,0.f,0.f,0.f};
#pragma unroll
      for (int kc = 0; kc < 2; ++kc) {
        bf16x16 a = lds_frag(ps + nt * 16 * PS + kc * 32, PS, lane);
        bf16x16 b = lds_frag(vt + dt * 16 * VTS + kc * 32, VTS, lane);
        acc = wmma_bf16(a, b, acc);
      }
#pragma unroll
      for (int rr = 0; rr < 8; ++rr) {
        const int tok = nt * 16 + ro + rr;
        outs[tok * XS + h * HDIM + dt * 16 + col] = f2bf(acc[rr]);
      }
    }
    __syncthreads();
  }

  // ---------------- output projection: 64x512 @ 512x512^T + b --------------
  float* orow = out + (size_t)w * NTOK * CDIM;
#pragma unroll 2
  for (int i = 0; i < 8; ++i) {
    const int t  = wave * 8 + i;
    const int rt = t >> 5;
    const int ot = t & 31;
    const int o0 = ot * 16;
    f32x8 acc = {0.f,0.f,0.f,0.f,0.f,0.f,0.f,0.f};
    const unsigned short* abase = outs + rt * 16 * XS;
    const unsigned short* wbase = proj_w + (size_t)o0 * CDIM;
#pragma unroll 4
    for (int kc = 0; kc < CDIM / 32; ++kc) {
      bf16x16 a = lds_frag(abase + kc * 32, XS, lane);
      bf16x16 b = gmem_wfrag(wbase + kc * 32, CDIM, lane);
      acc = wmma_bf16(a, b, acc);
    }
    const float pb = proj_b[o0 + col];
#pragma unroll
    for (int rr = 0; rr < 8; ++rr) {
      const int tok = rt * 16 + ro + rr;
      orow[(size_t)tok * CDIM + o0 + col] = acc[rr] + pb;
    }
  }
}

extern "C" void kernel_launch(void* const* d_in, const int* in_sizes, int n_in,
                              void* d_out, int out_size, void* d_ws, size_t ws_size,
                              hipStream_t stream) {
  (void)in_sizes; (void)n_in; (void)out_size; (void)ws_size;
  const float* x       = (const float*)d_in[0];
  const float* mask    = (const float*)d_in[1];
  const float* qkv_wf  = (const float*)d_in[2];
  const float* qkv_b   = (const float*)d_in[3];
  const float* proj_wf = (const float*)d_in[4];
  const float* proj_b  = (const float*)d_in[5];
  const float* btab    = (const float*)d_in[6];
  const int*   relidx  = (const int*)d_in[7];
  float* out = (float*)d_out;

  // bf16 weight staging in workspace (L2-resident: 1.5 MB + 0.5 MB)
  unsigned short* qkv_wb  = (unsigned short*)d_ws;
  unsigned short* proj_wb = qkv_wb + (size_t)3 * CDIM * CDIM;

  const int n4_qkv  = (3 * CDIM * CDIM) / 4;   // 196608
  const int n4_proj = (CDIM * CDIM) / 4;       // 65536
  hipLaunchKernelGGL(convert_w_bf16, dim3((n4_qkv + 255) / 256), dim3(256),
                     0, stream, qkv_wf, qkv_wb, n4_qkv);
  hipLaunchKernelGGL(convert_w_bf16, dim3((n4_proj + 255) / 256), dim3(256),
                     0, stream, proj_wf, proj_wb, n4_proj);

  dim3 grid(NWIN), block(NTHREADS);
  hipLaunchKernelGGL(swin_window_attention, grid, block, 0, stream,
                     x, mask, qkv_wb, qkv_b, proj_wb, proj_b, btab, relidx, out);
}
